// GlobalSceneTokenizer_70171175682082
// MI455X (gfx1250) — compile-verified, matrix operand order checked by reading, not run on hardware
//
#include <hip/hip_runtime.h>
#include <hip/hip_bf16.h>
#include <cstdint>

// ---------------- problem constants ----------------
#define NPTS   4096
#define NB     2
#define KNN    32
#define STEMC  64
#define TOKC   128
#define CIN    192          // 187 real channels padded to 192 (6 chunks of 32)
#define CINR   187
#define LSTR   208          // LDS row stride in f16 (pad 192 -> 208 to spread banks)
#define R2     0.0256f      // 0.16^2
#define ROWS_TOTAL (NB*NPTS*KNN)   // 262144
#define ROWS_PER_BLK 128

// Permuted input-channel layout (W1 rows permuted to match):
//   [0,64)    grouped point feature (f16, 16B aligned)
//   [64,160)  abs positional embedding (f16, 16B aligned at byte 128)
//   [160,163) relative xyz
//   [163,187) relative PE (3 axes x (4 sin + 4 cos))
//   [187,192) zero pad

typedef __attribute__((ext_vector_type(8)))  _Float16 v8h_t;
typedef __attribute__((ext_vector_type(16))) _Float16 v16h_t;
typedef __attribute__((ext_vector_type(8)))  float    v8f_t;
typedef __attribute__((ext_vector_type(4)))  int      v4i_t;

union Frag16 { v16h_t v; v8h_t h[2]; };

#if defined(__HIP_DEVICE_COMPILE__) && defined(__has_builtin)
#if __has_builtin(__builtin_amdgcn_global_load_async_to_lds_b128)
#define HAVE_ASYNC_LDS 1
#endif
#endif

#ifdef HAVE_ASYNC_LDS
typedef __attribute__((address_space(1))) v4i_t* gv4_t;   // global src (v4i*)
typedef __attribute__((address_space(3))) v4i_t* lv4_t;   // LDS dst (v4i*)
__device__ __forceinline__ void async_cp16(const void* g, void* l) {
    __builtin_amdgcn_global_load_async_to_lds_b128((gv4_t)g, (lv4_t)l, 0, 0);
}
__device__ __forceinline__ void wait_async0() {
#if __has_builtin(__builtin_amdgcn_s_wait_asynccnt)
    __builtin_amdgcn_s_wait_asynccnt(0);
#else
    asm volatile("s_wait_asynccnt 0x0" ::: "memory");
#endif
}
#endif

__device__ __forceinline__ unsigned fenc(float x) {
    unsigned u = __float_as_uint(x);
    return (u & 0x80000000u) ? ~u : (u | 0x80000000u);
}
__device__ __forceinline__ float fdec(unsigned e) {
    return __uint_as_float((e & 0x80000000u) ? (e & 0x7fffffffu) : ~e);
}

// ---------------- prep kernels ----------------

// W1 [187][128] f32 -> W1T [128][192] f16, rows permuted into the layout above
__global__ void gst_cvt_w1(const float* __restrict__ w1, _Float16* __restrict__ w1t) {
    int t = blockIdx.x * blockDim.x + threadIdx.x;      // 128*192
    if (t >= TOKC * CIN) return;
    int o = t / CIN, i = t % CIN;
    int src;
    if      (i < 64)  src = i;               // grouped feature   (orig [0,64))
    else if (i < 160) src = 91 + (i - 64);   // abs_pe            (orig [91,187))
    else if (i < 163) src = 64 + (i - 160);  // rel xyz           (orig [64,67))
    else if (i < 187) src = 67 + (i - 163);  // rel_pe            (orig [67,91))
    else              src = -1;              // pad
    float v = (src >= 0) ? w1[(size_t)src * TOKC + o] : 0.f;
    w1t[(size_t)o * CIN + i] = (_Float16)v;
}

// W2 [128][128] f32 -> W2T [128][128] f16
__global__ void gst_cvt_w2(const float* __restrict__ w2, _Float16* __restrict__ w2t) {
    int t = blockIdx.x * blockDim.x + threadIdx.x;
    if (t >= TOKC * TOKC) return;
    int o = t / TOKC, i = t % TOKC;
    w2t[(size_t)o * TOKC + i] = (_Float16)w2[(size_t)i * TOKC + o];
}

// point_feature f32 -> f16 (gathered K=32 times each; convert once)
__global__ void gst_cvt_pf(const float* __restrict__ pf, _Float16* __restrict__ pf16) {
    int t = blockIdx.x * blockDim.x + threadIdx.x;      // NB*NPTS*64
    if (t >= NB * NPTS * STEMC) return;
    pf16[t] = (_Float16)pf[t];
}

// abs positional embedding: per point 96 f16 (3 axes x [16 sin | 16 cos])
__global__ void gst_abspe(const float* __restrict__ xyz, _Float16* __restrict__ ape) {
    int t = blockIdx.x * blockDim.x + threadIdx.x;      // NB*NPTS*96
    if (t >= NB * NPTS * 96) return;
    int p = t / 96, c = t % 96;
    int axis = c / 32, r = c % 32;
    int i = (r < 16) ? r : r - 16;
    float freq = __expf(-0.6140226914650789f * (float)i);   // log(1e4)/15
    float ang = xyz[(size_t)p * 3 + axis] * freq;
    float v = (r < 16) ? __sinf(ang) : __cosf(ang);
    ape[t] = (_Float16)v;
}

__global__ void gst_init_gmax(unsigned* __restrict__ gmax) {
    if (threadIdx.x < NB * TOKC) gmax[threadIdx.x] = 0u;  // enc lower bound
}

// ---------------- ball query: one wave32 per query point ----------------
__global__ __launch_bounds__(256) void gst_ballq(const float* __restrict__ xyz,
                                                 int* __restrict__ nidx) {
    int wave = blockIdx.x * 8 + (threadIdx.x >> 5);     // NB*NPTS waves
    int lane = threadIdx.x & 31;
    int b = wave >> 12;            // /4096
    int n = wave & (NPTS - 1);
    const float* base = xyz + (size_t)b * NPTS * 3;
    float qx = base[n * 3 + 0], qy = base[n * 3 + 1], qz = base[n * 3 + 2];
    int out0 = (b * NPTS + n) * KNN;
    int count = 0, first = n;
    for (int s = 0; s < NPTS && count < KNN; s += 32) {
        int j = s + lane;
        float dx = base[j * 3 + 0] - qx;
        float dy = base[j * 3 + 1] - qy;
        float dz = base[j * 3 + 2] - qz;
        bool in = (dx * dx + dy * dy + dz * dz) <= R2;
        unsigned mask = (unsigned)__ballot(in);
        if (count == 0 && mask) first = s + __ffs(mask) - 1;
        if (in) {
            int slot = count + __popc(mask & ((1u << lane) - 1u));
            if (slot < KNN) nidx[out0 + slot] = j;
        }
        count += __popc(mask);
    }
    if (count > KNN) count = KNN;
    if (lane >= count) nidx[out0 + lane] = first;   // fill with first candidate
}

// ---------------- fused gather + PE + 2-layer MLP (WMMA) + max-reduce ----------------
__global__ __launch_bounds__(256, 2) void gst_mlp(
    const float* __restrict__ xyz, const _Float16* __restrict__ pf16,
    const int* __restrict__ nidx,
    const _Float16* __restrict__ w1t, const _Float16* __restrict__ w2t,
    const float* __restrict__ b1, const float* __restrict__ b2,
    const _Float16* __restrict__ ape, unsigned* __restrict__ gmax) {

    __shared__ _Float16 lds[ROWS_PER_BLK * LSTR];   // 53248 B
    __shared__ unsigned ldsmaxu[TOKC];

    const int tid = threadIdx.x;
    const int bb = blockIdx.x >> 10;                 // batch (1024 blocks per batch)

    if (tid == 0) { __builtin_prefetch(w1t, 0, 1); __builtin_prefetch(w2t, 0, 1); }

    // ---- build phase: 2 threads per row ----
    {
        int r = tid >> 1, half = tid & 1;
        int row = blockIdx.x * ROWS_PER_BLK + r;
        int b = row >> 17;                 // /(NPTS*KNN)
        int nk = row & (NPTS * KNN - 1);
        int n = nk >> 5;
        int j = nidx[(b << 17) + nk];
        _Float16* dst = &lds[r * LSTR];
        if (half == 0) {
            // grouped point feature f16: channels [0,64) = 128 B, 16B-aligned
            const _Float16* src = pf16 + (size_t)(b * NPTS + j) * STEMC;
#ifdef HAVE_ASYNC_LDS
            #pragma unroll
            for (int c = 0; c < 8; ++c) async_cp16(src + c * 8, dst + c * 8);
#else
            #pragma unroll
            for (int c = 0; c < 8; ++c)
                ((v8h_t*)dst)[c] = ((const v8h_t*)src)[c];
#endif
        } else {
            // abs PE: channels [64,160) = 192 B, 16B-aligned
            const _Float16* s = ape + (size_t)(b * NPTS + n) * 96;
#ifdef HAVE_ASYNC_LDS
            #pragma unroll
            for (int c = 0; c < 12; ++c) async_cp16(s + c * 8, dst + 64 + c * 8);
#else
            #pragma unroll
            for (int c = 0; c < 12; ++c)
                *(v8h_t*)(dst + 64 + c * 8) = *(const v8h_t*)(s + c * 8);
#endif
            const float* pj = xyz + (size_t)(b * NPTS + j) * 3;
            const float* pn = xyz + (size_t)(b * NPTS + n) * 3;
            float rel[3];
            #pragma unroll
            for (int a = 0; a < 3; ++a) {
                rel[a] = pj[a] - pn[a];
                dst[160 + a] = (_Float16)rel[a];   // channels [160,163)
            }
            // relative PE, channels [163,187): per axis 4 sin + 4 cos
            const float fr24[4] = {1.0f, 4.6415888336127775e-2f,
                                   2.1544346900318843e-3f, 1.0e-4f};
            #pragma unroll
            for (int a = 0; a < 3; ++a) {
                #pragma unroll
                for (int i = 0; i < 4; ++i) {
                    float ang = rel[a] * fr24[i];
                    dst[163 + a * 8 + i]     = (_Float16)__sinf(ang);
                    dst[163 + a * 8 + 4 + i] = (_Float16)__cosf(ang);
                }
            }
            // zero pad [187,192)
            #pragma unroll
            for (int c = CINR; c < CIN; ++c) dst[c] = (_Float16)0.f;
        }
        if (tid < TOKC) ldsmaxu[tid] = 0u;
    }
#ifdef HAVE_ASYNC_LDS
    wait_async0();            // this wave's async fills target only its own rows
#endif
    __syncthreads();

    // ---- matmul phase: one wave per 16-row strip ----
    const int wv = tid >> 5, lane = tid & 31;
    const int rbase = wv * 16;
    const int mrow = rbase + (lane & 15);
    const int k0 = (lane < 16) ? 0 : 8;          // 16-bit A/B fragment K origin
    const int nsub = lane & 15;                   // fragment column within tile
    const int moff = (lane < 16) ? 0 : 8;         // C/D row offset per lane half

    v8f_t acc[8];
    #pragma unroll
    for (int t = 0; t < 8; ++t) acc[t] = (v8f_t)(0.f);

    // layer 1: 192 = 6 chunks of K=32
    #pragma unroll
    for (int ch = 0; ch < 6; ++ch) {
        Frag16 a;
        const _Float16* ap = &lds[mrow * LSTR + ch * 32 + k0];
        a.h[0] = *(const v8h_t*)(ap);
        a.h[1] = *(const v8h_t*)(ap + 16);
        #pragma unroll
        for (int t = 0; t < 8; ++t) {
            Frag16 bf;
            const _Float16* wp = w1t + (size_t)(t * 16 + nsub) * CIN + ch * 32 + k0;
            bf.h[0] = *(const v8h_t*)(wp);
            bf.h[1] = *(const v8h_t*)(wp + 16);
            acc[t] = __builtin_amdgcn_wmma_f32_16x16x32_f16(
                false, a.v, false, bf.v, (short)0, acc[t], false, false);
        }
    }

    // bias + relu, write h1 (f16) back into this wave's own LDS rows, cols [0,128)
    #pragma unroll
    for (int t = 0; t < 8; ++t) {
        int col = t * 16 + nsub;
        float bias = b1[col];
        #pragma unroll
        for (int v = 0; v < 8; ++v) {
            float h = acc[t][v] + bias;
            h = fmaxf(h, 0.f);
            lds[(rbase + moff + v) * LSTR + col] = (_Float16)h;
        }
    }

    // layer 2: 128 = 4 chunks of K=32 (wave-private rows, no block barrier needed)
    v8f_t acc2[8];
    #pragma unroll
    for (int t = 0; t < 8; ++t) acc2[t] = (v8f_t)(0.f);
    #pragma unroll
    for (int ch = 0; ch < 4; ++ch) {
        Frag16 a;
        const _Float16* ap = &lds[mrow * LSTR + ch * 32 + k0];
        a.h[0] = *(const v8h_t*)(ap);
        a.h[1] = *(const v8h_t*)(ap + 16);
        #pragma unroll
        for (int t = 0; t < 8; ++t) {
            Frag16 bf;
            const _Float16* wp = w2t + (size_t)(t * 16 + nsub) * TOKC + ch * 32 + k0;
            bf.h[0] = *(const v8h_t*)(wp);
            bf.h[1] = *(const v8h_t*)(wp + 16);
            acc2[t] = __builtin_amdgcn_wmma_f32_16x16x32_f16(
                false, a.v, false, bf.v, (short)0, acc2[t], false, false);
        }
    }

    // bias + column-max over this wave's 16 rows, reduce into LDS, then global
    #pragma unroll
    for (int t = 0; t < 8; ++t) {
        int col = t * 16 + nsub;
        float bias = b2[col];
        float m = -3.4e38f;
        #pragma unroll
        for (int v = 0; v < 8; ++v) m = fmaxf(m, acc2[t][v] + bias);
        m = fmaxf(m, __shfl_xor(m, 16));      // merge the two lane halves (same col)
        if (lane < 16) atomicMax(&ldsmaxu[col], fenc(m));
    }
    __syncthreads();
    if (tid < TOKC) atomicMax(&gmax[bb * TOKC + tid], ldsmaxu[tid]);
}

// ---------------- tiny head: g = relu(g W3 + b3) W4 + b4 ----------------
__global__ __launch_bounds__(256) void gst_head(const unsigned* __restrict__ gmax,
                                                const float* __restrict__ w3,
                                                const float* __restrict__ b3,
                                                const float* __restrict__ w4,
                                                const float* __restrict__ b4,
                                                float* __restrict__ out) {
    __shared__ float g[NB][TOKC];
    __shared__ float h[NB][TOKC];
    int tid = threadIdx.x;                 // 256
    int b = tid >> 7, o = tid & 127;
    g[b][o] = fdec(gmax[tid]);
    __syncthreads();
    float s = b3[o];
    #pragma unroll 8
    for (int i = 0; i < TOKC; ++i) s += g[b][i] * w3[i * TOKC + o];
    h[b][o] = fmaxf(s, 0.f);
    __syncthreads();
    float s2 = b4[o];
    #pragma unroll 8
    for (int i = 0; i < TOKC; ++i) s2 += h[b][i] * w4[i * TOKC + o];
    out[tid] = s2;
}

// ---------------- launcher ----------------
extern "C" void kernel_launch(void* const* d_in, const int* in_sizes, int n_in,
                              void* d_out, int out_size, void* d_ws, size_t ws_size,
                              hipStream_t stream) {
    (void)in_sizes; (void)n_in; (void)out_size; (void)ws_size;
    const float* xyz = (const float*)d_in[0];
    const float* pf  = (const float*)d_in[1];
    const float* W1  = (const float*)d_in[2];
    const float* b1  = (const float*)d_in[3];
    const float* W2  = (const float*)d_in[4];
    const float* b2  = (const float*)d_in[5];
    const float* W3  = (const float*)d_in[6];
    const float* b3  = (const float*)d_in[7];
    const float* W4  = (const float*)d_in[8];
    const float* b4  = (const float*)d_in[9];
    float* out = (float*)d_out;

    char* ws = (char*)d_ws;
    size_t off = 0;
    int*       nidx = (int*)(ws + off);       off += (size_t)NB * NPTS * KNN * 4;          // 1 MB
    _Float16*  w1t  = (_Float16*)(ws + off);  off += (size_t)TOKC * CIN * 2;               // 48 KB
    _Float16*  w2t  = (_Float16*)(ws + off);  off += (size_t)TOKC * TOKC * 2;              // 32 KB
    _Float16*  ape  = (_Float16*)(ws + off);  off += (size_t)NB * NPTS * 96 * 2;           // 1.5 MB
    _Float16*  pf16 = (_Float16*)(ws + off);  off += (size_t)NB * NPTS * STEMC * 2;        // 1 MB
    unsigned*  gmax = (unsigned*)(ws + off);  off += (size_t)NB * TOKC * 4;                // 1 KB

    gst_cvt_w1<<<(TOKC * CIN + 255) / 256, 256, 0, stream>>>(W1, w1t);
    gst_cvt_w2<<<(TOKC * TOKC + 255) / 256, 256, 0, stream>>>(W2, w2t);
    gst_cvt_pf<<<(NB * NPTS * STEMC + 255) / 256, 256, 0, stream>>>(pf, pf16);
    gst_abspe<<<(NB * NPTS * 96 + 255) / 256, 256, 0, stream>>>(xyz, ape);
    gst_init_gmax<<<1, 256, 0, stream>>>(gmax);
    gst_ballq<<<(NB * NPTS) / 8, 256, 0, stream>>>(xyz, nidx);
    gst_mlp<<<ROWS_TOTAL / ROWS_PER_BLK, 256, 0, stream>>>(
        xyz, pf16, nidx, w1t, w2t, b1, b2, ape, gmax);
    gst_head<<<1, 256, 0, stream>>>(gmax, W3, b3, W4, b4, out);
}
